// Net_5901285064753
// MI455X (gfx1250) — compile-verified, matrix operand order checked by reading, not run on hardware
//
#include <hip/hip_runtime.h>
#include <hip/hip_bf16.h>
#include <math.h>

// Problem constants (from reference)
#define G_  4096
#define L_  64
#define E_  256
#define D_  128
#define H2_ 64

typedef __attribute__((ext_vector_type(16))) __bf16 v16bf;
typedef __attribute__((ext_vector_type(8)))  __bf16 v8bf;
typedef __attribute__((ext_vector_type(8)))  float  v8f;

union FragU { v16bf v; v8bf h[2]; };

// Load a WMMA 16-bit operand fragment from a row-major [*, stride] bf16 buffer.
// Per CDNA5 ISA layout: lanes 0-15 hold K = k0+{0..7} (VGPR0-3) and k0+16+{0..7}
// (VGPR4-7) for row `row`; lanes 16-31 hold the +8 K halves. Both runs are
// 16-byte contiguous -> two 16B loads per lane.
__device__ __forceinline__ v16bf load_frag(const __bf16* base, int stride, int row,
                                           int k0, int lane) {
  const int half = (lane >> 4) & 1;
  FragU u;
  u.h[0] = *(const v8bf*)(base + row * stride + k0 + half * 8);
  u.h[1] = *(const v8bf*)(base + row * stride + k0 + 16 + half * 8);
  return u.v;
}

__device__ __forceinline__ v8f wmma_bf16(v16bf a, v16bf b, v8f c) {
  // (neg_a, A, neg_b, B, c_mod, C, reuse_a, reuse_b)
  return __builtin_amdgcn_wmma_f32_16x16x32_bf16(false, a, false, b, (short)0, c,
                                                 false, false);
}

// ---------------------------------------------------------------------------
// Prep kernels
// ---------------------------------------------------------------------------

// out[n*K + k] = bf16(W[k*N + n])  (transpose f32 [K,N] -> bf16 [N,K])
__global__ void transpose_bf16(const float* __restrict__ W, __bf16* __restrict__ out,
                               int K, int N) {
  int i = blockIdx.x * blockDim.x + threadIdx.x;
  if (i < K * N) {
    int k = i / N, n = i % N;
    out[(size_t)n * K + k] = (__bf16)W[(size_t)k * N + n];
  }
}

// wn = w / ||w||  (128-wide)
__global__ void normalize_w(const float* __restrict__ w, float* __restrict__ wn) {
  __shared__ float red[D_];
  int t = threadIdx.x;
  float v = w[t];
  red[t] = v * v;
  __syncthreads();
  for (int s = D_ / 2; s > 0; s >>= 1) {
    if (t < s) red[t] += red[t + s];
    __syncthreads();
  }
  wn[t] = v * rsqrtf(red[0]);
}

// ---------------------------------------------------------------------------
// Fused per-graph GNN: 3 x (SAGE -> ReLU -> TopK pool -> mean pool accum)
// One block per graph, 256 threads (8 waves).
// Tile mapping: wave handles one 16-row M block (m0 = (wave&3)*16) and a
// 64-column N half (nb = (wave>>2)*64) -> 4 N tiles share the A fragments.
// ---------------------------------------------------------------------------
__global__ __launch_bounds__(256)
void gnn_fused(const int* __restrict__ x_idx, const int* __restrict__ edges,
               const unsigned char* __restrict__ node_mask_in,
               const unsigned char* __restrict__ edge_mask_in,
               const float* __restrict__ emb, const __bf16* __restrict__ wT,
               const float* __restrict__ bl1, const float* __restrict__ bl2,
               const float* __restrict__ bl3, const float* __restrict__ wn,
               float* __restrict__ z_out) {
  __shared__ __align__(16) float  xs[L_ * D_];     // current node features (f32)
  __shared__ __align__(16) __bf16 xb[L_ * D_];     // bf16 copy of xs (WMMA A)
  __shared__ __align__(16) __bf16 aggb[L_ * D_];   // bf16 mean-aggregated (WMMA A)
  __shared__ float score[L_];
  __shared__ int   nmask[L_];
  __shared__ int   keepA[L_];
  __shared__ int   emask[E_];
  __shared__ int   esrc[E_];
  __shared__ int   edst[E_];
  __shared__ float zsum[D_];
  __shared__ int   sh_k, sh_kept;

  const int g = blockIdx.x;
  const int tid = threadIdx.x;
  const int wave = tid >> 5;
  const int lane = tid & 31;
  const int lr = lane & 15;
  const int half = lane >> 4;

  // --- init masks / edges / zsum ---
  if (tid < L_) nmask[tid] = node_mask_in[g * L_ + tid] ? 1 : 0;
  if (tid < E_) {
    emask[tid] = edge_mask_in[g * E_ + tid] ? 1 : 0;
    esrc[tid] = edges[((size_t)g * E_ + tid) * 2 + 0];
    edst[tid] = edges[((size_t)g * E_ + tid) * 2 + 1];
  }
  if (tid < D_) zsum[tid] = 0.f;

  // --- embedding gather: xs = emb[x_idx[g]] ---
  for (int i = tid; i < L_ * (D_ / 4); i += 256) {
    int row = i / (D_ / 4), q = i % (D_ / 4);
    int v = x_idx[g * L_ + row];
    ((float4*)xs)[row * (D_ / 4) + q] = ((const float4*)(emb + (size_t)v * D_))[q];
  }
  __syncthreads();

  const float* bls[3] = {bl1, bl2, bl3};

  for (int layer = 0; layer < 3; ++layer) {
    // 1) xb = bf16(xs)
    for (int i = tid; i < L_ * D_; i += 256) xb[i] = (__bf16)xs[i];
    __syncthreads();

    // 2) deterministic per-node gather mean: aggb[m] = mean_{e: dst==m} xs[src]
    {
      const int m = tid >> 2;            // node (4 threads per node)
      const int f0 = (tid & 3) * 32;     // 32-feature slice
      float acc[32];
#pragma unroll
      for (int ff = 0; ff < 32; ++ff) acc[ff] = 0.f;
      float cm = 0.f;
      for (int e = 0; e < E_; ++e) {
        if (emask[e] && edst[e] == m) {
          const float* xr = &xs[esrc[e] * D_ + f0];
#pragma unroll
          for (int ff = 0; ff < 32; ++ff) acc[ff] += xr[ff];
          cm += 1.f;
        }
      }
      const float inv = 1.f / (cm > 1.f ? cm : 1.f);
#pragma unroll
      for (int ff = 0; ff < 32; ++ff)
        aggb[m * D_ + f0 + ff] = (__bf16)(acc[ff] * inv);
    }
    __syncthreads();

    // 3) WMMA: out = aggb @ Wl + xb @ Wr + bl ; ReLU ; node-mask ; -> xs
    {
      const __bf16* WlT = wT + ((size_t)layer * 2 + 0) * D_ * D_;  // [N=128][K=128]
      const __bf16* WrT = wT + ((size_t)layer * 2 + 1) * D_ * D_;
      const float* bl = bls[layer];
      const int m0 = (wave & 3) * 16;   // M block for this wave
      const int nb = (wave >> 2) * 64;  // N half for this wave (4 tiles)
      v8f acc[4] = {};
      for (int k0 = 0; k0 < D_; k0 += 32) {
        const v16bf aA = load_frag(aggb, D_, m0 + lr, k0, lane);
        const v16bf aX = load_frag(xb, D_, m0 + lr, k0, lane);
#pragma unroll
        for (int j = 0; j < 4; ++j) {
          const int n0 = nb + j * 16;
          v16bf bA = load_frag(WlT, D_, n0 + lr, k0, lane);
          acc[j] = wmma_bf16(aA, bA, acc[j]);
          v16bf bX = load_frag(WrT, D_, n0 + lr, k0, lane);
          acc[j] = wmma_bf16(aX, bX, acc[j]);
        }
      }
#pragma unroll
      for (int j = 0; j < 4; ++j) {
        const int n = nb + j * 16 + lr;
        const float bias = bl[n];
#pragma unroll
        for (int e2 = 0; e2 < 8; ++e2) {
          const int m = m0 + half * 8 + e2;  // C/D layout: vgpr e2 -> row m0+half*8+e2
          float v = acc[j][e2] + bias;
          v = v > 0.f ? v : 0.f;
          xs[m * D_ + n] = nmask[m] ? v : 0.f;
        }
      }
    }
    __syncthreads();

    // 4) TopK scoring: score = tanh(h . wn)
    const float* w = wn + layer * D_;
    if (tid < L_) {
      float s = 0.f;
      for (int k = 0; k < D_; ++k) s += xs[tid * D_ + k] * w[k];
      score[tid] = tanhf(s);
    }
    if (tid == 0) {
      int n = 0;
      for (int j = 0; j < L_; ++j) n += nmask[j];
      sh_k = (int)ceilf(0.8f * (float)n);        // matches jnp f32 semantics
    }
    __syncthreads();

    // 5) stable descending rank (ties -> lower index first), keep = rank < k
    if (tid < L_) {
      const float NEG = -3.402823466e38f;
      const float st = nmask[tid] ? score[tid] : NEG;
      int r = 0;
      for (int j = 0; j < L_; ++j) {
        const float sj = nmask[j] ? score[j] : NEG;
        if (sj > st || (sj == st && j < tid)) ++r;
      }
      keepA[tid] = (r < sh_k) && nmask[tid];
    }
    __syncthreads();

    // 6) apply pooling: scale kept rows by score, zero dropped; prune edges
    for (int i = tid; i < L_ * D_; i += 256) {
      const int m = i >> 7;  // / D_
      xs[i] = keepA[m] ? xs[i] * score[m] : 0.f;
    }
    if (tid < E_) emask[tid] = emask[tid] && keepA[esrc[tid]] && keepA[edst[tid]];
    if (tid < L_) nmask[tid] = keepA[tid];
    if (tid == 0) {
      int c = 0;
      for (int j = 0; j < L_; ++j) c += keepA[j];
      sh_kept = c > 1 ? c : 1;
    }
    __syncthreads();

    // 7) global mean pool accumulate
    if (tid < D_) {
      float s = 0.f;
      for (int m = 0; m < L_; ++m) s += xs[m * D_ + tid];
      zsum[tid] += s / (float)sh_kept;
    }
    __syncthreads();
  }

  if (tid < D_) z_out[(size_t)g * D_ + tid] = zsum[tid];
}

// ---------------------------------------------------------------------------
// MLP head: sigmoid(relu(relu(z@W1+b1)@W2+b2)@W3+b3). 64 graphs per block.
// ---------------------------------------------------------------------------
__global__ __launch_bounds__(256)
void mlp_head(const float* __restrict__ z, const __bf16* __restrict__ W1T,
              const float* __restrict__ b1, const __bf16* __restrict__ W2T,
              const float* __restrict__ b2, const float* __restrict__ W3,
              const float* __restrict__ b3, float* __restrict__ out) {
  __shared__ __align__(16) __bf16 zb[64 * D_];
  __shared__ __align__(16) __bf16 h1b[64 * D_];
  __shared__ float h2[64 * H2_];

  const int g0 = blockIdx.x * 64;
  const int tid = threadIdx.x;
  const int wave = tid >> 5;
  const int lane = tid & 31;
  const int lr = lane & 15;
  const int half = lane >> 4;

  for (int i = tid; i < 64 * D_; i += 256) zb[i] = (__bf16)z[(size_t)g0 * D_ + i];
  __syncthreads();

  // GEMM1: [64x128] = zb @ W1 (W1T is [128][128]); A reused across 4 N tiles
  {
    const int m0 = (wave & 3) * 16;
    const int nb = (wave >> 2) * 64;
    v8f acc[4] = {};
    for (int k0 = 0; k0 < D_; k0 += 32) {
      const v16bf a = load_frag(zb, D_, m0 + lr, k0, lane);
#pragma unroll
      for (int j = 0; j < 4; ++j) {
        v16bf b = load_frag(W1T, D_, nb + j * 16 + lr, k0, lane);
        acc[j] = wmma_bf16(a, b, acc[j]);
      }
    }
#pragma unroll
    for (int j = 0; j < 4; ++j) {
      const int n = nb + j * 16 + lr;
      const float bias = b1[n];
#pragma unroll
      for (int e2 = 0; e2 < 8; ++e2) {
        const int m = m0 + half * 8 + e2;
        float v = acc[j][e2] + bias;
        v = v > 0.f ? v : 0.f;
        h1b[m * D_ + n] = (__bf16)v;
      }
    }
  }
  __syncthreads();

  // GEMM2: [64x64] = h1 @ W2 (W2T is [64][128]); A reused across 2 N tiles
  {
    const int m0 = (wave & 3) * 16;
    const int nb = (wave >> 2) * 32;
    v8f acc[2] = {};
    for (int k0 = 0; k0 < D_; k0 += 32) {
      const v16bf a = load_frag(h1b, D_, m0 + lr, k0, lane);
#pragma unroll
      for (int j = 0; j < 2; ++j) {
        v16bf b = load_frag(W2T, D_, nb + j * 16 + lr, k0, lane);
        acc[j] = wmma_bf16(a, b, acc[j]);
      }
    }
#pragma unroll
    for (int j = 0; j < 2; ++j) {
      const int n = nb + j * 16 + lr;
      const float bias = b2[n];
#pragma unroll
      for (int e2 = 0; e2 < 8; ++e2) {
        const int m = m0 + half * 8 + e2;
        float v = acc[j][e2] + bias;
        v = v > 0.f ? v : 0.f;
        h2[m * H2_ + n] = v;
      }
    }
  }
  __syncthreads();

  if (tid < 64) {
    float s = b3[0];
    for (int k = 0; k < H2_; ++k) s += h2[tid * H2_ + k] * W3[k];
    out[g0 + tid] = 1.f / (1.f + expf(-s));
  }
}

// ---------------------------------------------------------------------------
extern "C" void kernel_launch(void* const* d_in, const int* in_sizes, int n_in,
                              void* d_out, int out_size, void* d_ws, size_t ws_size,
                              hipStream_t stream) {
  (void)in_sizes; (void)n_in; (void)out_size; (void)ws_size;
  const int* x_idx = (const int*)d_in[0];
  const int* edges = (const int*)d_in[1];
  const unsigned char* node_mask = (const unsigned char*)d_in[2];
  const unsigned char* edge_mask = (const unsigned char*)d_in[3];
  const float* emb = (const float*)d_in[4];
  const float* Wl[3] = {(const float*)d_in[5], (const float*)d_in[9],  (const float*)d_in[13]};
  const float* bl[3] = {(const float*)d_in[6], (const float*)d_in[10], (const float*)d_in[14]};
  const float* Wr[3] = {(const float*)d_in[7], (const float*)d_in[11], (const float*)d_in[15]};
  const float* wp[3] = {(const float*)d_in[8], (const float*)d_in[12], (const float*)d_in[16]};
  const float* W1 = (const float*)d_in[17];
  const float* b1 = (const float*)d_in[18];
  const float* W2 = (const float*)d_in[19];
  const float* b2 = (const float*)d_in[20];
  const float* W3 = (const float*)d_in[21];
  const float* b3 = (const float*)d_in[22];
  float* out = (float*)d_out;

  // workspace layout
  __bf16* wT  = (__bf16*)d_ws;            // 6 * 128*128 bf16 (WlT/WrT per layer)
  __bf16* W1T = wT + 6 * D_ * D_;         // 128*128 bf16
  __bf16* W2T = W1T + D_ * D_;            // 64*128 bf16
  float* wn = (float*)((char*)d_ws +
                       (size_t)(6 * D_ * D_ + D_ * D_ + H2_ * D_) * sizeof(__bf16));
  float* z = wn + 3 * D_;                 // [G, D] f32

  for (int c = 0; c < 3; ++c) {
    transpose_bf16<<<(D_ * D_ + 255) / 256, 256, 0, stream>>>(
        Wl[c], wT + (size_t)(2 * c + 0) * D_ * D_, D_, D_);
    transpose_bf16<<<(D_ * D_ + 255) / 256, 256, 0, stream>>>(
        Wr[c], wT + (size_t)(2 * c + 1) * D_ * D_, D_, D_);
    normalize_w<<<1, D_, 0, stream>>>(wp[c], wn + c * D_);
  }
  transpose_bf16<<<(D_ * D_ + 255) / 256, 256, 0, stream>>>(W1, W1T, D_, D_);
  transpose_bf16<<<(D_ * H2_ + 255) / 256, 256, 0, stream>>>(W2, W2T, D_, H2_);

  gnn_fused<<<G_, 256, 0, stream>>>(x_idx, edges, node_mask, edge_mask, emb, wT,
                                    bl[0], bl[1], bl[2], wn, z);
  mlp_head<<<G_ / 64, 256, 0, stream>>>(z, W1T, b1, W2T, b2, W3, b3, out);
}
